// DeformableAttentionFuser_5935644803542
// MI455X (gfx1250) — matrix operand order repaired and checked by reading, not compiled, not run on hardware
//
#include <hip/hip_runtime.h>
#include <hip/hip_bf16.h>

// ---------------------------------------------------------------------------
// Deformable attention fuser for MI455X (gfx1250, wave32, WMMA).
// B=2 L=2 H=100 W=100 C=256 NH=4 NL=1 NP=8 D=64 -> bs=4, Nq=10000, M=40000.
// GEMMs use V_WMMA_F32_16X16X32_BF16 with bf16x3 error compensation
// (a_hi*b_hi + a_hi*b_lo + a_lo*b_hi) for ~fp32 accuracy at bf16 rates.
// ---------------------------------------------------------------------------

typedef __attribute__((ext_vector_type(16))) __bf16 v16bf;
typedef __attribute__((ext_vector_type(8)))  __bf16 v8bf;
typedef __attribute__((ext_vector_type(8)))  float  v8f;

#define NQ      10000
#define GW      100
#define GH      100
#define CDIM    256
#define NHD     4
#define DDIM    64
#define NPTS    8
#define MROWS   40000          // bs*Nq
#define N1      352            // 256 (val) + 64 (off) + 32 (attn)
#define LDSS    264            // 256 + 8 bf16 pad -> conflict-free ds_load_b128

__device__ __forceinline__ v8f wmma_bf16(v16bf a, v16bf b, v8f c) {
  return __builtin_amdgcn_wmma_f32_16x16x32_bf16(false, a, false, b, (short)0, c,
                                                 false, false);
}

// Core 16x16 tile: A from LDS (hi/lo, [16][LDSS]), B from global ([n][k], k-major).
// Returns fp32 accumulator in the ISA C/D layout (VGPR r: M=r+8*khalf, N=lane&15).
__device__ __forceinline__ v8f gemm_core(const __bf16* __restrict__ sAh,
                                         const __bf16* __restrict__ sAl,
                                         const __bf16* __restrict__ Bh,
                                         const __bf16* __restrict__ Bl,
                                         int lane) {
  const int arow  = lane & 15;
  const int khalf = lane >> 4;
  const __bf16* ah = sAh + arow * LDSS;
  const __bf16* al = sAl + arow * LDSS;
  v8f a0 = {0,0,0,0,0,0,0,0};
  v8f a1 = {0,0,0,0,0,0,0,0};
  v8f a2 = {0,0,0,0,0,0,0,0};
#pragma unroll
  for (int ks = 0; ks < 8; ++ks) {
    const int kb = ks * 32;
    // A frag: elems j = K[kb + 16*khalf + j] (matches B)
    v8bf h0 = *(const v8bf*)(ah + kb + 16 * khalf);
    v8bf h1 = *(const v8bf*)(ah + kb + 16 * khalf + 8);
    v8bf l0 = *(const v8bf*)(al + kb + 16 * khalf);
    v8bf l1 = *(const v8bf*)(al + kb + 16 * khalf + 8);
    v16bf ahi = __builtin_shufflevector(h0, h1, 0,1,2,3,4,5,6,7,8,9,10,11,12,13,14,15);
    v16bf alo = __builtin_shufflevector(l0, l1, 0,1,2,3,4,5,6,7,8,9,10,11,12,13,14,15);
    // B frag: elems j = K[kb + 16*khalf + j] of column (lane&15); contiguous 32B
    v16bf bhi = *(const v16bf*)(Bh + kb + 16 * khalf);
    v16bf blo = *(const v16bf*)(Bl + kb + 16 * khalf);
    a0 = wmma_bf16(ahi, bhi, a0);
    a1 = wmma_bf16(ahi, blo, a1);
    a2 = wmma_bf16(alo, bhi, a2);
  }
  return a0 + a1 + a2;
}

// Stage one 16x256 fp32 A tile into LDS as bf16 hi/lo (256 threads, 16 elems each).
__device__ __forceinline__ void stage_A(const float* __restrict__ src_base,
                                        __bf16* __restrict__ sAh,
                                        __bf16* __restrict__ sAl, int tid) {
  const int ar  = tid >> 4;
  const int ac0 = (tid & 15) << 4;
  const float* xr = src_base + (size_t)ar * CDIM + ac0;
  __bf16* dh = sAh + ar * LDSS + ac0;
  __bf16* dl = sAl + ar * LDSS + ac0;
#pragma unroll
  for (int j = 0; j < 16; ++j) {
    float v  = xr[j];
    __bf16 h = (__bf16)v;
    dh[j] = h;
    dl[j] = (__bf16)(v - (float)h);
  }
}

// ---------------------------------------------------------------------------
// Kernel 0: split + transpose weights into bf16 hi/lo, [n][k] layout.
// ---------------------------------------------------------------------------
__global__ void __launch_bounds__(256)
pack_weights(const float* __restrict__ W_val, const float* __restrict__ W_off,
             const float* __restrict__ W_attn, const float* __restrict__ W_out,
             __bf16* __restrict__ w1h, __bf16* __restrict__ w1l,
             __bf16* __restrict__ w2h, __bf16* __restrict__ w2l) {
  int i = blockIdx.x * 256 + threadIdx.x;
  if (i < N1 * CDIM) {
    int n = i / CDIM, k = i % CDIM;
    float w = (n < 256) ? W_val[(size_t)k * 256 + n]
            : (n < 320) ? W_off[(size_t)k * 64 + (n - 256)]
                        : W_attn[(size_t)k * 32 + (n - 320)];
    __bf16 h = (__bf16)w;
    w1h[i] = h;
    w1l[i] = (__bf16)(w - (float)h);
  } else {
    int j = i - N1 * CDIM;          // < 256*256
    int n = j / CDIM, k = j % CDIM;
    float w = W_out[(size_t)k * 256 + n];
    __bf16 h = (__bf16)w;
    w2h[j] = h;
    w2l[j] = (__bf16)(w - (float)h);
  }
}

// ---------------------------------------------------------------------------
// Kernel 1: fused projection GEMM  x(40000x256) @ [W_val|W_off|W_attn](256x352)
// value scattered to (b, h, q, d) layout; off/attn logits row-major.
// grid = (2500, 3), 256 threads = 8 waves, one 16x16 N-tile per wave.
// ---------------------------------------------------------------------------
__global__ void __launch_bounds__(256)
gemm_proj(const float* __restrict__ x,
          const __bf16* __restrict__ w1h, const __bf16* __restrict__ w1l,
          const float* __restrict__ b_val, const float* __restrict__ b_off,
          const float* __restrict__ b_attn,
          float* __restrict__ value, float* __restrict__ offb,
          float* __restrict__ attnb) {
  __shared__ __bf16 sAh[16 * LDSS];
  __shared__ __bf16 sAl[16 * LDSS];
  const int tid   = threadIdx.x;
  const int mtile = blockIdx.x;
  stage_A(x + (size_t)mtile * 16 * CDIM, sAh, sAl, tid);
  __syncthreads();

  const int wave  = tid >> 5, lane = tid & 31;
  const int ntile = blockIdx.y * 8 + wave;
  if (ntile >= (N1 / 16)) return;

  const int col   = lane & 15;
  const int khalf = lane >> 4;
  const int n     = ntile * 16 + col;
  const size_t bo = (size_t)n * CDIM;
  v8f acc = gemm_core(sAh, sAl, w1h + bo, w1l + bo, lane);

  // 10000 % 16 == 0 -> whole M-tile lives in a single batch element.
  const int rowbase = mtile * 16 + khalf * 8;
  const int b  = rowbase / NQ;
  const int q0 = rowbase - b * NQ;

  if (n < 256) {
    const int hh = n >> 6, d = n & 63;
    const float bias = b_val[n];
    float* vp = value + (((size_t)(b * NHD + hh) * NQ + q0) * DDIM + d);
#pragma unroll
    for (int r = 0; r < 8; ++r) vp[(size_t)r * DDIM] = acc[r] + bias;
  } else if (n < 320) {
    const int j = n - 256;
    const float bias = b_off[j];
    float* op = offb + ((size_t)rowbase * 64 + j);
#pragma unroll
    for (int r = 0; r < 8; ++r) op[(size_t)r * 64] = acc[r] + bias;
  } else {
    const int j = n - 320;
    const float bias = b_attn[j];
    float* ap = attnb + ((size_t)rowbase * 32 + j);
#pragma unroll
    for (int r = 0; r < 8; ++r) ap[(size_t)r * 32] = acc[r] + bias;
  }
}

// ---------------------------------------------------------------------------
// Kernel 2: softmax + bilinear gather + attention combine.
// One wave per (b,q,h); each lane owns channels {2*lane, 2*lane+1} of D=64.
// value (41 MB) is L2-resident, so the 4-corner gathers run at L2 rates.
// grid = 160000/8 = 20000 blocks of 256.
// ---------------------------------------------------------------------------
__global__ void __launch_bounds__(256)
deform_sample(const float* __restrict__ value, const float* __restrict__ offb,
              const float* __restrict__ attnb, float* __restrict__ mid) {
  const int tid  = threadIdx.x;
  const int wave = tid >> 5, lane = tid & 31;
  const int idx  = blockIdx.x * 8 + wave;   // over bs*Nq*NH, h fastest
  const int hh   = idx & 3;
  const int rq   = idx >> 2;                // b*Nq + q
  const int q    = rq % NQ;
  const int b    = rq / NQ;

  const float* offr = offb  + (size_t)rq * 64 + hh * (NPTS * 2);
  const float* attr = attnb + (size_t)rq * 32 + hh * NPTS;

  // softmax over the 8 sampling points (all lanes redundantly; broadcast loads)
  float lg[NPTS];
  float mx = -3.0e38f;
#pragma unroll
  for (int p = 0; p < NPTS; ++p) { lg[p] = attr[p]; mx = fmaxf(mx, lg[p]); }
  float s = 0.f;
#pragma unroll
  for (int p = 0; p < NPTS; ++p) { lg[p] = __expf(lg[p] - mx); s += lg[p]; }
  const float inv = 1.0f / s;

  const int qx = q % GW, qy = q / GW;
  const float* vb = value + (size_t)(b * NHD + hh) * NQ * DDIM + lane * 2;

  float accx = 0.f, accy = 0.f;
#pragma unroll
  for (int p = 0; p < NPTS; ++p) {
    // loc*W - 0.5 with ref=(q+0.5)/W collapses to q + offset
    const float X = (float)qx + offr[2 * p + 0];
    const float Y = (float)qy + offr[2 * p + 1];
    const float x0f = floorf(X), y0f = floorf(Y);
    const float lx = X - x0f, ly = Y - y0f;
    const int x0 = (int)x0f, y0 = (int)y0f;
    const float wp = lg[p] * inv;
#pragma unroll
    for (int dy = 0; dy < 2; ++dy) {
#pragma unroll
      for (int dx = 0; dx < 2; ++dx) {
        const int xi = x0 + dx, yi = y0 + dy;
        if (xi >= 0 && xi < GW && yi >= 0 && yi < GH) {   // zeros padding
          const float wgt = (dx ? lx : 1.f - lx) * (dy ? ly : 1.f - ly) * wp;
          const float2 g = *(const float2*)(vb + (size_t)(yi * GW + xi) * DDIM);
          accx += wgt * g.x;
          accy += wgt * g.y;
        }
      }
    }
  }
  float2 o = {accx, accy};
  *(float2*)(mid + (size_t)rq * CDIM + hh * DDIM + lane * 2) = o;
}

// ---------------------------------------------------------------------------
// Kernel 3: output GEMM  mid(40000x256) @ W_out(256x256) + b_out -> d_out.
// grid = (2500, 2), 8 waves * 2 = all 16 N-tiles.
// ---------------------------------------------------------------------------
__global__ void __launch_bounds__(256)
gemm_out(const float* __restrict__ mid,
         const __bf16* __restrict__ w2h, const __bf16* __restrict__ w2l,
         const float* __restrict__ b_out, float* __restrict__ out) {
  __shared__ __bf16 sAh[16 * LDSS];
  __shared__ __bf16 sAl[16 * LDSS];
  const int tid   = threadIdx.x;
  const int mtile = blockIdx.x;
  stage_A(mid + (size_t)mtile * 16 * CDIM, sAh, sAl, tid);
  __syncthreads();

  const int wave  = tid >> 5, lane = tid & 31;
  const int ntile = blockIdx.y * 8 + wave;   // 0..15, always valid
  const int col   = lane & 15;
  const int khalf = lane >> 4;
  const int n     = ntile * 16 + col;
  const size_t bo = (size_t)n * CDIM;
  v8f acc = gemm_core(sAh, sAl, w2h + bo, w2l + bo, lane);

  const int rowbase = mtile * 16 + khalf * 8;
  const float bias  = b_out[n];
  float* op = out + ((size_t)rowbase * CDIM + n);
#pragma unroll
  for (int r = 0; r < 8; ++r) op[(size_t)r * CDIM] = acc[r] + bias;
}

// ---------------------------------------------------------------------------
// Launch. Workspace layout (bytes, all 256-aligned):
//   value  [4*4*10000*64 f32]   @ 0          (40,960,000)
//   mid    [40000*256 f32]      @ 40,960,000 (40,960,000)
//   offb   [40000*64 f32]       @ 81,920,000 (10,240,000)
//   attnb  [40000*32 f32]       @ 92,160,000 ( 5,120,000)
//   w1h/w1l[352*256 bf16]       @ 97,280,000 / 97,460,224
//   w2h/w2l[256*256 bf16]       @ 97,640,448 / 97,771,520   (end ~93.4 MiB)
// ---------------------------------------------------------------------------
extern "C" void kernel_launch(void* const* d_in, const int* in_sizes, int n_in,
                              void* d_out, int out_size, void* d_ws, size_t ws_size,
                              hipStream_t stream) {
  const float* inputs = (const float*)d_in[0];
  const float* W_off  = (const float*)d_in[1];
  const float* b_off  = (const float*)d_in[2];
  const float* W_attn = (const float*)d_in[3];
  const float* b_attn = (const float*)d_in[4];
  const float* W_val  = (const float*)d_in[5];
  const float* b_val  = (const float*)d_in[6];
  const float* W_out  = (const float*)d_in[7];
  const float* b_out  = (const float*)d_in[8];
  float* out = (float*)d_out;

  char* ws = (char*)d_ws;
  float*  value = (float*)(ws + 0);
  float*  mid   = (float*)(ws + 40960000);
  float*  offb  = (float*)(ws + 81920000);
  float*  attnb = (float*)(ws + 92160000);
  __bf16* w1h   = (__bf16*)(ws + 97280000);
  __bf16* w1l   = (__bf16*)(ws + 97460224);
  __bf16* w2h   = (__bf16*)(ws + 97640448);
  __bf16* w2l   = (__bf16*)(ws + 97771520);

  pack_weights<<<608, 256, 0, stream>>>(W_val, W_off, W_attn, W_out,
                                        w1h, w1l, w2h, w2l);
  gemm_proj<<<dim3(MROWS / 16, 3), 256, 0, stream>>>(inputs, w1h, w1l,
                                                     b_val, b_off, b_attn,
                                                     value, offb, attnb);
  deform_sample<<<(MROWS * NHD) / 8, 256, 0, stream>>>(value, offb, attnb, mid);
  gemm_out<<<dim3(MROWS / 16, 2), 256, 0, stream>>>(mid, w2h, w2l, b_out, out);
}
